// RNN_88648124991536
// MI455X (gfx1250) — matrix operand order repaired and checked by reading, not compiled, not run on hardware
//
#include <hip/hip_runtime.h>
#include <hip/hip_bf16.h>

// ---------------------------------------------------------------------------
// CDNA5 (gfx1250) wave32 WMMA implementation of the GRU-scan + phase MLP.
// bf16 16x16x32 WMMA, f32 accumulate. Weights pre-transposed to [N][K] bf16.
// B_tile = 32 rows per WG: halves L2 weight traffic vs 16, one B-fragment
// feeds two row-tile WMMAs. All K-loops fully unrolled via template KX.
// ---------------------------------------------------------------------------

typedef __attribute__((ext_vector_type(16))) __bf16 v16bf;
typedef __attribute__((ext_vector_type(8)))  float  v8f;

union Frag { v16bf v; unsigned int u[8]; };

// A-matrix fragment (16x32 bf16), source row-major [16][ldk].
// Layout (matches B): lanes 0-15 -> K 0..15; lanes 16-31 -> K 16..31.
__device__ __forceinline__ v16bf load_a_frag(const __bf16* base, int ldk, int kk) {
  const int lane = threadIdx.x & 31;
  const int row  = lane & 15;
  const int kb   = (lane >> 4) ? 16 : 0;
  const unsigned int* p = (const unsigned int*)(base + row * ldk + kk + kb);
  Frag f;
#pragma unroll
  for (int i = 0; i < 4; ++i) f.u[i] = p[i];
#pragma unroll
  for (int i = 0; i < 4; ++i) f.u[4 + i] = p[4 + i];   // +8 halves
  return f.v;
}

// B-matrix fragment (32x16 bf16), weights stored transposed [N][K]:
// lane holds column n; lanes 0-15 -> K 0..15, lanes 16-31 -> K 16..31
// (16 contiguous halves = 32 bytes per lane).
__device__ __forceinline__ v16bf load_b_frag(const __bf16* wrow_kk) {
  const int lane = threadIdx.x & 31;
  const unsigned int* p = (const unsigned int*)(wrow_kk + ((lane >> 4) ? 16 : 0));
  Frag f;
#pragma unroll
  for (int i = 0; i < 8; ++i) f.u[i] = p[i];
  return f.v;
}

__device__ __forceinline__ v8f wmma_bf16(v16bf a, v16bf b, v8f c) {
  return __builtin_amdgcn_wmma_f32_16x16x32_bf16(false, a, false, b, (short)0, c,
                                                 false, false);
}

__device__ __forceinline__ float sigmoidf_(float x) { return 1.f / (1.f + __expf(-x)); }
__device__ __forceinline__ float tanhf_(float x)    { float e = __expf(2.f * x); return (e - 1.f) / (e + 1.f); }

// ---------------------------------------------------------------------------
// Prep: dst[N][Kpad] bf16 = transpose(src[K][N] f32), zero-padded in K.
// ---------------------------------------------------------------------------
__global__ void conv_T_kernel(const float* __restrict__ src, __bf16* __restrict__ dst,
                              int K, int N, int Kpad) {
  int i = blockIdx.x * blockDim.x + threadIdx.x;
  int total = N * Kpad;
  if (i >= total) return;
  int n = i / Kpad, k = i % Kpad;
  float v = (k < K) ? src[(size_t)k * N + n] : 0.f;
  dst[i] = (__bf16)v;
}

// ---------------------------------------------------------------------------
// GRU layer building blocks (B_tile = 32 rows => 2 row-tiles of 16).
// ---------------------------------------------------------------------------
#define BT    32                // batch rows per workgroup
#define HROW  (BT * 256)        // elements in one hidden buffer

// matmul phase: 64 (m,j) tiles; each produces rows 0-15 (acc0) and 16-31 (acc1)
template <int KX>
__device__ __forceinline__ void gru_matmul_phase(
    const __bf16* __restrict__ Xb, const __bf16* __restrict__ Hb,
    const __bf16* __restrict__ WiT, const __bf16* __restrict__ WhT,
    const float* __restrict__ bi0, const float* __restrict__ bi1,
    const float* __restrict__ bi2, const float* __restrict__ bh,
    __bf16* __restrict__ sScr, int wave, int lane) {
  for (int tl = wave; tl < 64; tl += 8) {
    const int m = tl >> 4, j = tl & 15;
    const int n = (j << 4) + (lane & 15);
    v8f acc0 = {}, acc1 = {};
    if (m < 3) {  // input-side contribution (gates r, z, n)
      const __bf16* wrow = WiT + ((size_t)m * 256 + n) * KX;
      __builtin_prefetch(wrow, 0, 1);   // global_prefetch_b8
#pragma unroll
      for (int kk = 0; kk < KX; kk += 32) {
        v16bf bf = load_b_frag(wrow + kk);
        acc0 = wmma_bf16(load_a_frag(Xb, KX, kk), bf, acc0);
        acc1 = wmma_bf16(load_a_frag(Xb + 16 * KX, KX, kk), bf, acc1);
      }
    }
    if (m != 2) {  // hidden-side contribution (gates r, z, n-hidden)
      const int gate = (m == 3) ? 2 : m;
      const __bf16* wrow = WhT + ((size_t)gate * 256 + n) * 256;
      __builtin_prefetch(wrow, 0, 1);
#pragma unroll
      for (int kk = 0; kk < 256; kk += 32) {
        v16bf bf = load_b_frag(wrow + kk);
        acc0 = wmma_bf16(load_a_frag(Hb, 256, kk), bf, acc0);
        acc1 = wmma_bf16(load_a_frag(Hb + 16 * 256, 256, kk), bf, acc1);
      }
    }
    const float bv = (m == 3) ? bh[n] : (m == 0 ? bi0[n] : (m == 1 ? bi1[n] : bi2[n]));
#pragma unroll
    for (int q = 0; q < 8; ++q) {
      const int M = q + ((lane >> 4) << 3);  // C/D layout: M = q (+8 hi-lanes)
      float v0 = acc0[q] + bv, v1 = acc1[q] + bv;
      if (m < 2) { v0 = sigmoidf_(v0); v1 = sigmoidf_(v1); }  // r, z
      sScr[m * HROW + M * 256 + n]        = (__bf16)v0;
      sScr[m * HROW + (16 + M) * 256 + n] = (__bf16)v1;
    }
  }
}

// elementwise GRU update: thread 'tid' owns column tid for all BT rows
__device__ __forceinline__ void gru_elementwise(const __bf16* __restrict__ sScr,
                                                __bf16* __restrict__ H, int tid) {
  for (int row = 0; row < BT; ++row) {
    const int i = row * 256 + tid;
    float r  = (float)sScr[0 * HROW + i];
    float z  = (float)sScr[1 * HROW + i];
    float ai = (float)sScr[2 * HROW + i];
    float ah = (float)sScr[3 * HROW + i];
    float h  = (float)H[i];
    float nn = tanhf_(ai + r * ah);
    H[i] = (__bf16)((1.f - z) * nn + z * h);
  }
}

// ---------------------------------------------------------------------------
// GRU scan kernel: one WG = 32 batch rows, 8 waves, T=64 sequential steps.
// ---------------------------------------------------------------------------
struct GruArgs {
  const int*    inp;        // [4096][128] occupations
  const __bf16* WiT[3];     // [3 gates][256][Kx] transposed bf16
  const __bf16* WhT[3];     // [3 gates][256][256]
  const float*  bi[3][3];   // [gru][gate r,z,n] input biases
  const float*  bh[3];      // bhn per gru
  const float*  W2;         // [256][4] f32
  const float*  b2;         // [4] f32
  float*        out;        // complex64 interleaved: out[2b] = real
};

__global__ __launch_bounds__(256) void gru_scan_kernel(GruArgs A) {
  __shared__ __bf16 sX[BT * 128];      // masked state (GRU1 input)
  __shared__ __bf16 sH[3][HROW];       // hidden states
  __shared__ __bf16 sScr[4 * HROW];    // r, z, a_in, a_hn pre-activations
  __shared__ float  sAccRow[BT];
  __shared__ float  sD[BT][4];

  const int tid  = threadIdx.x;
  const int lane = tid & 31;
  const int wave = tid >> 5;
  const int b0   = blockIdx.x * BT;

  for (int i = tid; i < BT * 128; i += 256) sX[i] = (__bf16)0.f;
  for (int i = tid; i < HROW; i += 256) {
    sH[0][i] = (__bf16)0.f; sH[1][i] = (__bf16)0.f; sH[2][i] = (__bf16)0.f;
  }
  if (tid < BT) sAccRow[tid] = 0.f;
  __syncthreads();

  for (int t = 0; t < 64; ++t) {
    // incremental autoregressive mask: step t reveals orbital t-1 (both spins)
    if (t > 0 && tid < BT) {
      int b = b0 + tid;
      sX[tid * 128 + (t - 1)]      = (__bf16)(float)A.inp[b * 128 + (t - 1)];
      sX[tid * 128 + 64 + (t - 1)] = (__bf16)(float)A.inp[b * 128 + 64 + (t - 1)];
    }
    __syncthreads();

    // ---- GRU1 (KX = 128) ----
    gru_matmul_phase<128>(sX, sH[0], A.WiT[0], A.WhT[0],
                          A.bi[0][0], A.bi[0][1], A.bi[0][2], A.bh[0],
                          sScr, wave, lane);
    __syncthreads();
    gru_elementwise(sScr, sH[0], tid);
    __syncthreads();

    // ---- GRU2 (KX = 256) ----
    gru_matmul_phase<256>(sH[0], sH[1], A.WiT[1], A.WhT[1],
                          A.bi[1][0], A.bi[1][1], A.bi[1][2], A.bh[1],
                          sScr, wave, lane);
    __syncthreads();
    gru_elementwise(sScr, sH[1], tid);
    __syncthreads();

    // ---- GRU3 (KX = 256) ----
    gru_matmul_phase<256>(sH[1], sH[2], A.WiT[2], A.WhT[2],
                          A.bi[2][0], A.bi[2][1], A.bi[2][2], A.bh[2],
                          sScr, wave, lane);
    __syncthreads();
    gru_elementwise(sScr, sH[2], tid);
    __syncthreads();

    // dense2 (256->4) on relu(h3), double log-normalize, gather idx, accumulate
    if (tid < BT * 4) {
      const int row = tid >> 2, c = tid & 3;
      float s = A.b2[c];
      for (int k = 0; k < 256; ++k) {
        float h = fmaxf((float)sH[2][row * 256 + k], 0.f);
        s += h * A.W2[k * 4 + c];
      }
      sD[row][c] = s;
    }
    __syncthreads();
    if (tid < BT) {
      float x[4];
#pragma unroll
      for (int c = 0; c < 4; ++c) x[c] = sD[tid][c];
#pragma unroll
      for (int it = 0; it < 2; ++it) {  // x -= logsumexp(2x)/2, twice
        float mx = fmaxf(fmaxf(x[0], x[1]), fmaxf(x[2], x[3]));
        float se = 0.f;
#pragma unroll
        for (int c = 0; c < 4; ++c) se += __expf(2.f * (x[c] - mx));
        float l = mx + 0.5f * __logf(se);
#pragma unroll
        for (int c = 0; c < 4; ++c) x[c] -= l;
      }
      int b   = b0 + tid;
      int idx = A.inp[b * 128 + t] + 2 * A.inp[b * 128 + 64 + t];
      sAccRow[tid] += x[idx];
    }
    __syncthreads();
  }

  if (tid < BT) A.out[2 * (b0 + tid)] = sAccRow[tid];  // real part
}

// ---------------------------------------------------------------------------
// Phase MLP: L0 (126->1024, K padded to 128), L1 (1024->1024), L2 scalar tail.
// ---------------------------------------------------------------------------
__global__ __launch_bounds__(256) void phase_l0_kernel(const int* __restrict__ inp,
                                                       const __bf16* __restrict__ WT,
                                                       const float* __restrict__ bias,
                                                       __bf16* __restrict__ actOut) {
  __shared__ __bf16 sA[16 * 128];
  const int tid = threadIdx.x, lane = tid & 31, wave = tid >> 5;
  const int b0 = blockIdx.x * 16;
  for (int i = tid; i < 16 * 128; i += 256) {
    int row = i >> 7, c = i & 127;
    float v = 0.f;
    if (c < 63)       v = (float)inp[(b0 + row) * 128 + c];
    else if (c < 126) v = (float)inp[(b0 + row) * 128 + c + 1];
    sA[i] = (__bf16)v;
  }
  __syncthreads();
  const int n = blockIdx.y * 128 + wave * 16 + (lane & 15);
  const __bf16* wrow = WT + (size_t)n * 128;
  v8f acc = {};
#pragma unroll
  for (int kk = 0; kk < 128; kk += 32)
    acc = wmma_bf16(load_a_frag(sA, 128, kk), load_b_frag(wrow + kk), acc);
  const float bv = bias[n];
#pragma unroll
  for (int q = 0; q < 8; ++q) {
    int M = q + ((lane >> 4) << 3);
    actOut[(size_t)(b0 + M) * 1024 + n] = (__bf16)fmaxf(acc[q] + bv, 0.f); // store relu'd
  }
}

__global__ __launch_bounds__(256) void phase_l1_kernel(const __bf16* __restrict__ actIn,
                                                       const __bf16* __restrict__ WT,
                                                       const float* __restrict__ bias,
                                                       __bf16* __restrict__ actOut) {
  const int tid = threadIdx.x, lane = tid & 31, wave = tid >> 5;
  const int b0 = blockIdx.x * 16;
  const int n  = blockIdx.y * 128 + wave * 16 + (lane & 15);
  const __bf16* Ab   = actIn + (size_t)b0 * 1024;
  const __bf16* wrow = WT + (size_t)n * 1024;
  __builtin_prefetch(wrow, 0, 1);
  v8f acc = {};
#pragma unroll
  for (int kk = 0; kk < 1024; kk += 32)
    acc = wmma_bf16(load_a_frag(Ab, 1024, kk), load_b_frag(wrow + kk), acc);
  const float bv = bias[n];
#pragma unroll
  for (int q = 0; q < 8; ++q) {
    int M = q + ((lane >> 4) << 3);
    actOut[(size_t)(b0 + M) * 1024 + n] = (__bf16)fmaxf(acc[q] + bv, 0.f);
  }
}

__global__ __launch_bounds__(256) void phase_l2_kernel(const __bf16* __restrict__ actIn,
                                                       const __bf16* __restrict__ WT, // [4][1024]
                                                       const float* __restrict__ bias,
                                                       const int* __restrict__ inp,
                                                       float* __restrict__ out) {
  __shared__ float sDD[64][4];
  const int tid = threadIdx.x;
  const int b0  = blockIdx.x * 64;
  const int row = tid >> 2, c = tid & 3;
  const __bf16* a = actIn + (size_t)(b0 + row) * 1024;
  const __bf16* w = WT + (size_t)c * 1024;
  float s = bias[c];
  for (int k = 0; k < 1024; ++k) s += (float)a[k] * (float)w[k]; // actIn already relu'd
  sDD[row][c] = s;
  __syncthreads();
  if (tid < 64) {
    int b    = b0 + tid;
    int pidx = inp[b * 128 + 63] + 2 * inp[b * 128 + 127];
    out[2 * b + 1] = 3.14159265358979323846f * sDD[tid][pidx];  // imag part
  }
}

// ---------------------------------------------------------------------------
// Host
// ---------------------------------------------------------------------------
extern "C" void kernel_launch(void* const* d_in, const int* in_sizes, int n_in,
                              void* d_out, int out_size, void* d_ws, size_t ws_size,
                              hipStream_t stream) {
  (void)out_size; (void)ws_size; (void)n_in;
  // Detect flatten order: JAX sorted dict keys -> leaf1 is dense1.b (128);
  // insertion order -> leaf1 is dense1.w (16384).
  const bool sorted = (in_sizes[1] == 128);

  const int* inp = (const int*)d_in[0];
  const float *d2w, *d2b;
  const float *Wi[3][3], *Wh[3][3], *bi[3][3], *bh[3];
  const float *pw[3], *pb[3];

  if (sorted) {
    d2b = (const float*)d_in[3]; d2w = (const float*)d_in[4];
    for (int g = 0; g < 3; ++g) {
      int base = 5 + g * 10;  // Whn, Whr, Whz, Win, Wir, Wiz, bhn, bin, bir, biz
      Wh[g][2] = (const float*)d_in[base + 0];
      Wh[g][0] = (const float*)d_in[base + 1];
      Wh[g][1] = (const float*)d_in[base + 2];
      Wi[g][2] = (const float*)d_in[base + 3];
      Wi[g][0] = (const float*)d_in[base + 4];
      Wi[g][1] = (const float*)d_in[base + 5];
      bh[g]    = (const float*)d_in[base + 6];
      bi[g][2] = (const float*)d_in[base + 7];
      bi[g][0] = (const float*)d_in[base + 8];
      bi[g][1] = (const float*)d_in[base + 9];
    }
    pb[0] = (const float*)d_in[35]; pw[0] = (const float*)d_in[36];
    pb[1] = (const float*)d_in[37]; pw[1] = (const float*)d_in[38];
    pb[2] = (const float*)d_in[39]; pw[2] = (const float*)d_in[40];
  } else {
    for (int g = 0; g < 3; ++g) {
      int base = 3 + g * 10;  // Wir, Wiz, Win, Whr, Whz, Whn, bir, biz, bin, bhn
      Wi[g][0] = (const float*)d_in[base + 0];
      Wi[g][1] = (const float*)d_in[base + 1];
      Wi[g][2] = (const float*)d_in[base + 2];
      Wh[g][0] = (const float*)d_in[base + 3];
      Wh[g][1] = (const float*)d_in[base + 4];
      Wh[g][2] = (const float*)d_in[base + 5];
      bi[g][0] = (const float*)d_in[base + 6];
      bi[g][1] = (const float*)d_in[base + 7];
      bi[g][2] = (const float*)d_in[base + 8];
      bh[g]    = (const float*)d_in[base + 9];
    }
    d2w = (const float*)d_in[33]; d2b = (const float*)d_in[34];
    pw[0] = (const float*)d_in[35]; pb[0] = (const float*)d_in[36];
    pw[1] = (const float*)d_in[37]; pb[1] = (const float*)d_in[38];
    pw[2] = (const float*)d_in[39]; pb[2] = (const float*)d_in[40];
  }

  // workspace carve-out (bf16 halves); every region fully rewritten each call
  __bf16* ws = (__bf16*)d_ws;
  size_t off = 0;
  auto alloc = [&](size_t halves) { __bf16* p = ws + off; off += halves; return p; };
  __bf16* WiT[3]; __bf16* WhT[3];
  WiT[0] = alloc((size_t)3 * 256 * 128); WhT[0] = alloc((size_t)3 * 256 * 256);
  WiT[1] = alloc((size_t)3 * 256 * 256); WhT[1] = alloc((size_t)3 * 256 * 256);
  WiT[2] = alloc((size_t)3 * 256 * 256); WhT[2] = alloc((size_t)3 * 256 * 256);
  __bf16* P0T  = alloc((size_t)1024 * 128);
  __bf16* P1T  = alloc((size_t)1024 * 1024);
  __bf16* P2T  = alloc((size_t)4 * 1024);
  __bf16* actA = alloc((size_t)4096 * 1024);
  __bf16* actB = alloc((size_t)4096 * 1024);

  auto convT = [&](const float* src, __bf16* dst, int K, int N, int Kpad) {
    int total = N * Kpad;
    conv_T_kernel<<<(total + 255) / 256, 256, 0, stream>>>(src, dst, K, N, Kpad);
  };
  for (int g = 0; g < 3; ++g) {
    int Kx = (g == 0) ? 128 : 256;
    for (int m = 0; m < 3; ++m) {
      convT(Wi[g][m], WiT[g] + (size_t)m * 256 * Kx, Kx, 256, Kx);
      convT(Wh[g][m], WhT[g] + (size_t)m * 256 * 256, 256, 256, 256);
    }
  }
  convT(pw[0], P0T, 126, 1024, 128);
  convT(pw[1], P1T, 1024, 1024, 1024);
  convT(pw[2], P2T, 1024, 4, 1024);

  GruArgs A;
  A.inp = inp;
  for (int g = 0; g < 3; ++g) {
    A.WiT[g] = WiT[g]; A.WhT[g] = WhT[g]; A.bh[g] = bh[g];
    for (int m = 0; m < 3; ++m) A.bi[g][m] = bi[g][m];
  }
  A.W2 = d2w; A.b2 = d2b; A.out = (float*)d_out;
  gru_scan_kernel<<<4096 / BT, 256, 0, stream>>>(A);

  phase_l0_kernel<<<dim3(4096 / 16, 8), 256, 0, stream>>>(inp, P0T, pb[0], actA);
  phase_l1_kernel<<<dim3(4096 / 16, 8), 256, 0, stream>>>(actA, P1T, pb[1], actB);
  phase_l2_kernel<<<4096 / 64, 256, 0, stream>>>(actB, P2T, pb[2], inp, (float*)d_out);
}
